// TemporalDebertaAggregator_9594956939672
// MI455X (gfx1250) — matrix operand order repaired and checked by reading, not compiled, not run on hardware
//
#include <hip/hip_runtime.h>
#include <hip/hip_bf16.h>
#include <math.h>

// ---------------- model constants ----------------
#define BB     16
#define SS     511
#define S1     512
#define DD     512
#define HH     8
#define DHD    64
#define FFD    2048
#define LL     2
#define KSPAN  513          // att_span
#define P2K    (2*KSPAN)    // 1026
#define LNEPS  1e-7f
#define INV_SCALE 0.07216878364870322f   // 1/sqrt(64*3)

typedef __attribute__((ext_vector_type(16))) __bf16 v16bf;
typedef __attribute__((ext_vector_type(4)))  __bf16 v4bf;
typedef __attribute__((ext_vector_type(8)))  float  v8f;

__device__ __forceinline__ v8f wmma_bf16(v16bf a, v16bf b, v8f c) {
  // D = A(16x32 bf16) * B(32x16 bf16) + C(16x16 f32)
  return __builtin_amdgcn_wmma_f32_16x16x32_bf16(false, a, false, b, (short)0, c,
                                                 false, false);
}

// ---- direct-from-global fragment loaders (f32 rows, contiguous in k) ----
// A fragment 16x32: lane holds row (lane&15); K chunk [kb_a, kb_a+16),
// kb_a = (lane>>4)*16. Contiguous -> float4 x4.
__device__ __forceinline__ v16bf load_afrag_f32(const float* __restrict__ pr, int k0, int kb_a) {
  float4 f0 = *(const float4*)(pr + k0 + kb_a);
  float4 f1 = *(const float4*)(pr + k0 + kb_a + 4);
  float4 f2 = *(const float4*)(pr + k0 + kb_a + 8);
  float4 f3 = *(const float4*)(pr + k0 + kb_a + 12);
  v16bf a;
  a[0]  = (__bf16)f0.x; a[1]  = (__bf16)f0.y; a[2]  = (__bf16)f0.z; a[3]  = (__bf16)f0.w;
  a[4]  = (__bf16)f1.x; a[5]  = (__bf16)f1.y; a[6]  = (__bf16)f1.z; a[7]  = (__bf16)f1.w;
  a[8]  = (__bf16)f2.x; a[9]  = (__bf16)f2.y; a[10] = (__bf16)f2.z; a[11] = (__bf16)f2.w;
  a[12] = (__bf16)f3.x; a[13] = (__bf16)f3.y; a[14] = (__bf16)f3.z; a[15] = (__bf16)f3.w;
  return a;
}
// B fragment 32x16 where B[k][n] = row_n[k]: lane holds column n = lane&15;
// K chunk [kb_b, kb_b+16), kb_b = (lane>>4)*16. Contiguous -> float4 x4.
__device__ __forceinline__ v16bf load_bfrag_f32(const float* __restrict__ pr, int k0, int kb_b) {
  float4 f0 = *(const float4*)(pr + k0 + kb_b);
  float4 f1 = *(const float4*)(pr + k0 + kb_b + 4);
  float4 f2 = *(const float4*)(pr + k0 + kb_b + 8);
  float4 f3 = *(const float4*)(pr + k0 + kb_b + 12);
  v16bf b;
  b[0]  = (__bf16)f0.x; b[1]  = (__bf16)f0.y; b[2]  = (__bf16)f0.z; b[3]  = (__bf16)f0.w;
  b[4]  = (__bf16)f1.x; b[5]  = (__bf16)f1.y; b[6]  = (__bf16)f1.z; b[7]  = (__bf16)f1.w;
  b[8]  = (__bf16)f2.x; b[9]  = (__bf16)f2.y; b[10] = (__bf16)f2.z; b[11] = (__bf16)f2.w;
  b[12] = (__bf16)f3.x; b[13] = (__bf16)f3.y; b[14] = (__bf16)f3.z; b[15] = (__bf16)f3.w;
  return b;
}

// ---------------- generic batched WMMA GEMM, LDS-staged bf16 ----------------
// C[bz] = act(A[bz] @ B[bz] + bias), bz -> (b = bz/Hs, h = bz%Hs)
// Block = 256 threads (8 waves), macro-tile 64x64, K-step 32.
// Requires: N % 64 == 0, Kd % 32 == 0; M arbitrary (rows clamped+zeroed).
// act: 0 = none, 1 = exact GELU
__global__ void wmma_gemm_kernel(const float* __restrict__ A, const float* __restrict__ Bm,
                                 const float* __restrict__ bias, float* __restrict__ C,
                                 int M, int N, int Kd, int lda, int ldb, int ldc,
                                 unsigned long long sAb, unsigned long long sAh,
                                 unsigned long long sBb, unsigned long long sBh,
                                 unsigned long long sCb, unsigned long long sCh,
                                 int Hs, int act) {
  // pitch 40 bf16 = 80 B keeps every 16-byte fragment chunk aligned
  __shared__ __bf16 As[64][40];   // [row][k]
  __shared__ __bf16 Bs[64][40];   // [n][k]  (transposed during staging)

  const int bz = blockIdx.z;
  const int bb = bz / Hs, hh = bz % Hs;
  const float* Ab = A + (size_t)bb * sAb + (size_t)hh * sAh;
  const float* Bb = Bm + (size_t)bb * sBb + (size_t)hh * sBh;
  float* Cb = C + (size_t)bb * sCb + (size_t)hh * sCh;

  const int tid  = threadIdx.x;
  const int lane = tid & 31;
  const int w    = tid >> 5;
  const int m0   = blockIdx.y * 64;
  const int n0   = blockIdx.x * 64;

  // staging coordinates (fully coalesced float4 global loads)
  const int ar  = tid >> 3;              // 0..31 (A row within half)
  const int ac4 = (tid & 7) * 4;         // 0..28 (A k-col, step 4)
  const int bkr = tid >> 4;              // 0..15 (B k-row within half)
  const int bc4 = (tid & 15) * 4;        // 0..60 (B n-col, step 4)

  // per-wave output tile: rows mi*16..+15, cols nh..nh+31 (two 16x16 tiles)
  const int mi   = w >> 1;
  const int nh   = (w & 1) * 32;
  const int li   = lane & 15;
  const int kb_a = (lane >> 4) * 8;
  const int kb_b = (lane >> 4) * 16;

  v8f acc0 = {}, acc1 = {};

  for (int k0 = 0; k0 < Kd; k0 += 32) {
    // ---- stage A 64x32 (clamp rows, zero via multiply: no exec branches) ----
#pragma unroll
    for (int hr = 0; hr < 2; ++hr) {
      int row  = ar + hr * 32;
      int grow = m0 + row;
      int gr   = (grow < M) ? grow : (M - 1);
      float4 f = *(const float4*)(Ab + (size_t)gr * lda + k0 + ac4);
      float s  = (grow < M) ? 1.0f : 0.0f;
      v4bf pk;
      pk[0] = (__bf16)(f.x * s); pk[1] = (__bf16)(f.y * s);
      pk[2] = (__bf16)(f.z * s); pk[3] = (__bf16)(f.w * s);
      *(v4bf*)&As[row][ac4] = pk;
    }
    // ---- stage B 32x64, transposed into Bs[n][k] ----
#pragma unroll
    for (int hk = 0; hk < 2; ++hk) {
      int k = bkr + hk * 16;
      float4 f = *(const float4*)(Bb + (size_t)(k0 + k) * ldb + n0 + bc4);
      Bs[bc4 + 0][k] = (__bf16)f.x;
      Bs[bc4 + 1][k] = (__bf16)f.y;
      Bs[bc4 + 2][k] = (__bf16)f.z;
      Bs[bc4 + 3][k] = (__bf16)f.w;
    }
    // prefetch next k-slice into cache hierarchy (global_prefetch_b8)
    if (k0 + 32 < Kd) {
      int grow = m0 + ar;
      int gr   = (grow < M) ? grow : (M - 1);
      __builtin_prefetch(Ab + (size_t)gr * lda + k0 + 32 + ac4, 0, 3);
      __builtin_prefetch(Bb + (size_t)(k0 + 32 + bkr) * ldb + n0 + bc4, 0, 3);
    }
    __syncthreads();

    // ---- fragments straight out of LDS as contiguous b128 chunks ----
    union Frag { v16bf v; float4 q[2]; };
    Frag af, b0, b1;
    const __bf16* arow = &As[mi * 16 + li][0];
    af.q[0] = *(const float4*)(arow + kb_b);
    af.q[1] = *(const float4*)(arow + kb_b + 8);
    const __bf16* br0 = &Bs[nh + li][0];
    const __bf16* br1 = &Bs[nh + 16 + li][0];
    b0.q[0] = *(const float4*)(br0 + kb_b);
    b0.q[1] = *(const float4*)(br0 + kb_b + 8);
    b1.q[0] = *(const float4*)(br1 + kb_b);
    b1.q[1] = *(const float4*)(br1 + kb_b + 8);

    acc0 = wmma_bf16(af.v, b0.v, acc0);
    acc1 = wmma_bf16(af.v, b1.v, acc1);
    __syncthreads();
  }

  // ---- epilogue ----
  const int mbase = m0 + mi * 16 + (lane >> 4) * 8;
  const int nc0 = n0 + nh + li;
  const int nc1 = n0 + nh + 16 + li;
  const float bd0 = bias ? bias[nc0] : 0.0f;
  const float bd1 = bias ? bias[nc1] : 0.0f;
#pragma unroll
  for (int i = 0; i < 8; ++i) {
    int m = mbase + i;
    if (m < M) {
      float v0 = acc0[i] + bd0;
      float v1 = acc1[i] + bd1;
      if (act == 1) {
        v0 = 0.5f * v0 * (1.0f + erff(v0 * 0.70710678118654752f));
        v1 = 0.5f * v1 * (1.0f + erff(v1 * 0.70710678118654752f));
      }
      Cb[(size_t)m * ldc + nc0] = v0;
      Cb[(size_t)m * ldc + nc1] = v1;
    }
  }
}

// ---------------- fused disentangled score kernel ----------------
// scores[b,h,i,j] = (q_i.k_j + q_i.posk[i-j+K] + k_j.posq[i-j+K]) / sqrt(192), masked.
// One wave per 16x16 tile; banded position-bias via 16x32 WMMA + LDS diagonal gather.
__global__ void scores_kernel(const float* __restrict__ q, const float* __restrict__ km,
                              const float* __restrict__ posk, const float* __restrict__ posq,
                              const unsigned char* __restrict__ mask,
                              float* __restrict__ scores) {
  __shared__ float sS[4][16][16];
  __shared__ float sC[4][16][32];
  __shared__ float sP[4][16][32];

  const int lane = threadIdx.x & 31;
  const int w    = threadIdx.x >> 5;
  const int z    = blockIdx.z;              // b*H + h
  const int b    = z >> 3;
  const int hh   = z & 7;
  const int i0   = blockIdx.y * 16;
  const int j0   = (blockIdx.x * 4 + w) * 16;
  const int hcol = hh * DHD;
  const int li   = lane & 15;
  const int kb_a = (lane >> 4) * 8;
  const int kb_b = (lane >> 4) * 16;
  const int d    = i0 - j0 + KSPAN;         // center relative index of tile

  const float* qrow = q  + ((size_t)(b * S1 + i0 + li)) * DD + hcol;
  const float* krow = km + ((size_t)(b * S1 + j0 + li)) * DD + hcol;
  // position-table rows this lane needs (r = half*16 + li, p = d-15+r)
  int p0 = d - 15 + li;       if (p0 < 0) p0 = 0; if (p0 > P2K - 1) p0 = P2K - 1;
  int p1 = d + 1 + li;        if (p1 < 0) p1 = 0; if (p1 > P2K - 1) p1 = P2K - 1;
  const float* pk0 = posk + (size_t)p0 * DD + hcol;
  const float* pk1 = posk + (size_t)p1 * DD + hcol;
  const float* pq0 = posq + (size_t)p0 * DD + hcol;
  const float* pq1 = posq + (size_t)p1 * DD + hcol;

  v8f accS = {}, accC0 = {}, accC1 = {}, accP0 = {}, accP1 = {};
#pragma unroll
  for (int kk = 0; kk < 2; ++kk) {          // DH = 64 -> two K=32 steps
    const int k0 = kk * 32;
    v16bf aq  = load_afrag_f32(qrow, k0, kb_b);   // Q rows (i)
    v16bf ak  = load_afrag_f32(krow, k0, kb_b);   // K rows (j), as A for p2c
    v16bf bK  = load_bfrag_f32(krow, k0, kb_b);   // K^T for q.k
    v16bf bC0 = load_bfrag_f32(pk0,  k0, kb_b);   // posk band cols 0..15
    v16bf bC1 = load_bfrag_f32(pk1,  k0, kb_b);   // posk band cols 16..31
    v16bf bP0 = load_bfrag_f32(pq0,  k0, kb_b);
    v16bf bP1 = load_bfrag_f32(pq1,  k0, kb_b);
    accS  = wmma_bf16(aq, bK,  accS);
    accC0 = wmma_bf16(aq, bC0, accC0);
    accC1 = wmma_bf16(aq, bC1, accC1);
    accP0 = wmma_bf16(ak, bP0, accP0);
    accP1 = wmma_bf16(ak, bP1, accP1);
  }

  // spill D fragments to LDS (C/D layout: vgpr i -> M = i + 8*(lane>=16), N = lane&15)
#pragma unroll
  for (int i = 0; i < 8; ++i) {
    int mm = i + (lane >> 4) * 8;
    sS[w][mm][li]      = accS[i];
    sC[w][mm][li]      = accC0[i];
    sC[w][mm][16 + li] = accC1[i];
    sP[w][mm][li]      = accP0[i];
    sP[w][mm][16 + li] = accP1[i];
  }
  __syncthreads();

  // gather diagonals: element (ii,jj) uses band column r = 15 + ii - jj in [0,30]
#pragma unroll
  for (int t = 0; t < 8; ++t) {
    int e  = t * 32 + lane;
    int ii = e >> 4, jj = e & 15;
    int r  = 15 + ii - jj;
    float val = (sS[w][ii][jj] + sC[w][ii][r] + sP[w][jj][r]) * INV_SCALE;
    int gi = i0 + ii, gj = j0 + jj;
    float mi = (gi == 0) ? 1.0f : (float)mask[(size_t)b * SS + gi - 1];
    float mj = (gj == 0) ? 1.0f : (float)mask[(size_t)b * SS + gj - 1];
    if (mi * mj <= 0.0f) val = -3.38953139e38f;
    scores[(((size_t)z * S1) + gi) * S1 + gj] = val;
  }
}

// ---------------- softmax * ext (XSoftmax) ----------------
__global__ void softmax_kernel(float* __restrict__ probs, const unsigned char* __restrict__ mask) {
  __shared__ float red[256];
  const size_t row = blockIdx.x;          // z*S1 + i
  const int i = (int)(row & (S1 - 1));
  const int b = (int)(row >> 12);         // row / (H*S1)
  float* p = probs + row * (size_t)S1;
  const int t = threadIdx.x;

  float x0 = p[t], x1 = p[t + 256];
  red[t] = fmaxf(x0, x1);
  __syncthreads();
  for (int s = 128; s > 0; s >>= 1) { if (t < s) red[t] = fmaxf(red[t], red[t + s]); __syncthreads(); }
  float mx = red[0];
  __syncthreads();

  float e0 = __expf(x0 - mx), e1 = __expf(x1 - mx);
  red[t] = e0 + e1;
  __syncthreads();
  for (int s = 128; s > 0; s >>= 1) { if (t < s) red[t] += red[t + s]; __syncthreads(); }
  float inv = 1.0f / red[0];

  float mi  = (i == 0) ? 1.0f : (float)mask[(size_t)b * SS + i - 1];
  float mj0 = (t == 0) ? 1.0f : (float)mask[(size_t)b * SS + t - 1];
  float mj1 = (float)mask[(size_t)b * SS + t + 255];
  p[t]       = e0 * inv * mi * mj0;
  p[t + 256] = e1 * inv * mi * mj1;
}

// ---------------- residual add + LayerNorm (in-place on h) ----------------
__global__ void add_ln_kernel(const float* __restrict__ x, float* __restrict__ h,
                              const float* __restrict__ g, const float* __restrict__ bta) {
  __shared__ float red[256];
  const size_t row = blockIdx.x;
  const float* xr = x + row * (size_t)DD;
  float* hr = h + row * (size_t)DD;
  const int t = threadIdx.x;

  float v0 = xr[t] + hr[t], v1 = xr[t + 256] + hr[t + 256];
  red[t] = v0 + v1;
  __syncthreads();
  for (int s = 128; s > 0; s >>= 1) { if (t < s) red[t] += red[t + s]; __syncthreads(); }
  float mu = red[0] * (1.0f / DD);
  __syncthreads();
  float d0 = v0 - mu, d1 = v1 - mu;
  red[t] = d0 * d0 + d1 * d1;
  __syncthreads();
  for (int s = 128; s > 0; s >>= 1) { if (t < s) red[t] += red[t + s]; __syncthreads(); }
  float inv = rsqrtf(red[0] * (1.0f / DD) + LNEPS);
  hr[t]       = d0 * inv * g[t] + bta[t];
  hr[t + 256] = d1 * inv * g[t + 256] + bta[t + 256];
}

// ---------------- embed (prepend CLS) and extract ----------------
__global__ void embed_kernel(const float* __restrict__ x, const float* __restrict__ cls,
                             float* __restrict__ h) {
  size_t idx = (size_t)blockIdx.x * blockDim.x + threadIdx.x;
  if (idx >= (size_t)BB * S1 * DD) return;
  int dd = (int)(idx % DD);
  size_t r = idx / DD;
  int s1 = (int)(r % S1);
  int b  = (int)(r / S1);
  h[idx] = (s1 == 0) ? cls[dd] : x[((size_t)b * SS + (s1 - 1)) * DD + dd];
}

__global__ void extract_kernel(const float* __restrict__ h, float* __restrict__ out) {
  int idx = blockIdx.x * blockDim.x + threadIdx.x;
  if (idx >= BB * DD) return;
  int b = idx / DD, dd = idx % DD;
  out[idx] = h[((size_t)b * S1) * DD + dd];
}

// ---------------- host orchestration ----------------
static void launch_gemm(const float* A, const float* Bm, const float* bias, float* C,
                        int M, int N, int Kd, int lda, int ldb, int ldc,
                        unsigned long long sAb, unsigned long long sAh,
                        unsigned long long sBb, unsigned long long sBh,
                        unsigned long long sCb, unsigned long long sCh,
                        int batch, int Hs, int act, hipStream_t s) {
  dim3 grid((N + 63) / 64, (M + 63) / 64, batch);
  wmma_gemm_kernel<<<grid, 256, 0, s>>>(A, Bm, bias, C, M, N, Kd, lda, ldb, ldc,
                                        sAb, sAh, sBb, sBh, sCb, sCh, Hs, act);
}

extern "C" void kernel_launch(void* const* d_in, const int* in_sizes, int n_in,
                              void* d_out, int out_size, void* d_ws, size_t ws_size,
                              hipStream_t stream) {
  const float* x    = (const float*)d_in[0];
  const unsigned char* mask = (const unsigned char*)d_in[1];
  const float* cls  = (const float*)d_in[2];
  const float* rel  = (const float*)d_in[3];
  const float* Wq   = (const float*)d_in[4];
  const float* bq   = (const float*)d_in[5];
  const float* Wk   = (const float*)d_in[6];
  const float* bk   = (const float*)d_in[7];
  const float* Wv   = (const float*)d_in[8];
  const float* bv   = (const float*)d_in[9];
  const float* Wpq  = (const float*)d_in[10];
  const float* bpq  = (const float*)d_in[11];
  const float* Wpk  = (const float*)d_in[12];
  const float* bpk  = (const float*)d_in[13];
  const float* Wo   = (const float*)d_in[14];
  const float* bo   = (const float*)d_in[15];
  const float* ln1g = (const float*)d_in[16];
  const float* ln1b = (const float*)d_in[17];
  const float* Wi   = (const float*)d_in[18];
  const float* bi   = (const float*)d_in[19];
  const float* Wo2  = (const float*)d_in[20];
  const float* bo2  = (const float*)d_in[21];
  const float* ln2g = (const float*)d_in[22];
  const float* ln2b = (const float*)d_in[23];
  float* out = (float*)d_out;

  // workspace layout (floats)
  float* ws = (float*)d_ws;
  size_t off = 0;
  float* h    = ws + off; off += (size_t)BB * S1 * DD;        // 16 MB
  float* qb   = ws + off; off += (size_t)BB * S1 * DD;
  float* kb   = ws + off; off += (size_t)BB * S1 * DD;
  float* vb   = ws + off; off += (size_t)BB * S1 * DD;
  float* posk = ws + off; off += (size_t)P2K * DD;
  float* posq = ws + off; off += (size_t)P2K * DD;
  float* sc   = ws + off; off += (size_t)BB * HH * S1 * S1;   // 128 MB
  float* ctx  = ws + off; off += (size_t)BB * S1 * DD;
  float* tmp  = ws + off; off += (size_t)BB * S1 * DD;
  float* ffb  = ws + off; off += (size_t)BB * S1 * FFD;       // 64 MB
  (void)ws_size; (void)in_sizes; (void)n_in; (void)out_size;

  const int M = BB * S1;  // 8192 rows of activations

  embed_kernel<<<(unsigned)(((size_t)BB * S1 * DD + 255) / 256), 256, 0, stream>>>(x, cls, h);

  for (int l = 0; l < LL; ++l) {
    const float* Wq_l  = Wq  + (size_t)l * DD * DD;  const float* bq_l  = bq  + (size_t)l * DD;
    const float* Wk_l  = Wk  + (size_t)l * DD * DD;  const float* bk_l  = bk  + (size_t)l * DD;
    const float* Wv_l  = Wv  + (size_t)l * DD * DD;  const float* bv_l  = bv  + (size_t)l * DD;
    const float* Wpq_l = Wpq + (size_t)l * DD * DD;  const float* bpq_l = bpq + (size_t)l * DD;
    const float* Wpk_l = Wpk + (size_t)l * DD * DD;  const float* bpk_l = bpk + (size_t)l * DD;
    const float* Wo_l  = Wo  + (size_t)l * DD * DD;  const float* bo_l  = bo  + (size_t)l * DD;
    const float* Wi_l  = Wi  + (size_t)l * DD * FFD; const float* bi_l  = bi  + (size_t)l * FFD;
    const float* Wo2_l = Wo2 + (size_t)l * FFD * DD; const float* bo2_l = bo2 + (size_t)l * DD;

    // Q, K, V projections: [8192,512] @ [512,512]
    launch_gemm(h, Wq_l, bq_l, qb, M, DD, DD, DD, DD, DD, 0,0,0,0,0,0, 1, 1, 0, stream);
    launch_gemm(h, Wk_l, bk_l, kb, M, DD, DD, DD, DD, DD, 0,0,0,0,0,0, 1, 1, 0, stream);
    launch_gemm(h, Wv_l, bv_l, vb, M, DD, DD, DD, DD, DD, 0,0,0,0,0,0, 1, 1, 0, stream);
    // position projections: [1026,512] @ [512,512] (rows clamped+zeroed)
    launch_gemm(rel, Wpk_l, bpk_l, posk, P2K, DD, DD, DD, DD, DD, 0,0,0,0,0,0, 1, 1, 0, stream);
    launch_gemm(rel, Wpq_l, bpq_l, posq, P2K, DD, DD, DD, DD, DD, 0,0,0,0,0,0, 1, 1, 0, stream);

    // fused scores: QK^T + banded c2p/p2c + mask + scale
    scores_kernel<<<dim3(S1 / 64, S1 / 16, BB * HH), 128, 0, stream>>>(qb, kb, posk, posq, mask, sc);

    // XSoftmax
    softmax_kernel<<<BB * HH * S1, 256, 0, stream>>>(sc, mask);

    // ctx = probs @ V, batched over (b,h): [512,512] @ [512,64]
    launch_gemm(sc, vb, nullptr, ctx, S1, DHD, S1, S1, DD, DD,
                (unsigned long long)HH * S1 * S1, (unsigned long long)S1 * S1,
                (unsigned long long)S1 * DD, DHD,
                (unsigned long long)S1 * DD, DHD,
                BB * HH, HH, 0, stream);

    // attention out-proj + residual LN
    launch_gemm(ctx, Wo_l, bo_l, tmp, M, DD, DD, DD, DD, DD, 0,0,0,0,0,0, 1, 1, 0, stream);
    add_ln_kernel<<<M, 256, 0, stream>>>(tmp, h, ln1g + (size_t)l * DD, ln1b + (size_t)l * DD);

    // FFN: gelu(h @ Wi + bi) @ Wo2 + bo2, residual LN
    launch_gemm(h, Wi_l, bi_l, ffb, M, FFD, DD, DD, FFD, FFD, 0,0,0,0,0,0, 1, 1, 1, stream);
    launch_gemm(ffb, Wo2_l, bo2_l, tmp, M, DD, FFD, FFD, DD, DD, 0,0,0,0,0,0, 1, 1, 0, stream);
    add_ln_kernel<<<M, 256, 0, stream>>>(tmp, h, ln2g + (size_t)l * DD, ln2b + (size_t)l * DD);
  }

  extract_kernel<<<(BB * DD + 255) / 256, 256, 0, stream>>>(h, out);
}
